// GPSP_Autoencoder_24068996727355
// MI455X (gfx1250) — matrix operand order repaired and checked by reading, not compiled, not run on hardware
//
#include <hip/hip_runtime.h>

// ---------------------------------------------------------------------------
// Types for CDNA5 WMMA (wave32): 16x16x32 bf16 -> f32
// ---------------------------------------------------------------------------
typedef __attribute__((ext_vector_type(8)))  __bf16 v8bf;
typedef __attribute__((ext_vector_type(16))) __bf16 v16bf;
typedef __attribute__((ext_vector_type(8)))  float  v8f;

static constexpr int cN0 = 32768, cN1 = 8192, cN2 = 2048, cG = 8;
static constexpr int cE0 = 262144, cE1 = 65536, cE2 = 16384;

static inline int cdiv(int a, int b) { return (a + b - 1) / b; }

// ---------------------------------------------------------------------------
// Device helpers
// ---------------------------------------------------------------------------
__device__ __forceinline__ unsigned short f2bf(float f) {
  unsigned u = __float_as_uint(f);
  u += 0x7fffu + ((u >> 16) & 1u);           // round-to-nearest-even
  return (unsigned short)(u >> 16);
}
// order-preserving float<->uint encoding for atomicMax-based segment max
__device__ __forceinline__ unsigned fenc(float f) {
  unsigned u = __float_as_uint(f);
  return (u & 0x80000000u) ? ~u : (u | 0x80000000u);
}
__device__ __forceinline__ float fdec(unsigned e) {
  unsigned u = (e & 0x80000000u) ? (e & 0x7fffffffu) : ~e;
  return __uint_as_float(u);
}

// ---------------------------------------------------------------------------
// fp32 -> bf16 conversion kernels (with K padding to multiple of 32)
// ---------------------------------------------------------------------------
__global__ void cvt_pad_bf16(const float* __restrict__ src, unsigned short* __restrict__ dst,
                             int n, int K, int Kp) {
  int i = blockIdx.x * 256 + threadIdx.x;
  if (i >= n * Kp) return;
  int r = i / Kp, k = i - r * Kp;
  float v = (k < K) ? src[(size_t)r * K + k] : 0.f;
  dst[i] = f2bf(v);
}

// Wl[K,C], Wr[K,C]  ->  dst[2C][Kp] column-major-transposed (row j = output col)
__global__ void cvt_wcat_t(const float* __restrict__ Wl, const float* __restrict__ Wr,
                           unsigned short* __restrict__ dst, int K, int C, int Kp) {
  int i = blockIdx.x * 256 + threadIdx.x;
  int total = 2 * C * Kp;
  if (i >= total) return;
  int j = i / Kp, k = i - j * Kp;
  float v = 0.f;
  if (k < K) v = (j < C) ? Wl[(size_t)k * C + j] : Wr[(size_t)k * C + (j - C)];
  dst[i] = f2bf(v);
}

// W[K,N] -> dst[N][Kp]
__global__ void cvt_w_t(const float* __restrict__ W, unsigned short* __restrict__ dst,
                        int K, int N, int Kp) {
  int i = blockIdx.x * 256 + threadIdx.x;
  if (i >= N * Kp) return;
  int j = i / Kp, k = i - j * Kp;
  float v = (k < K) ? W[(size_t)k * N + j] : 0.f;
  dst[i] = f2bf(v);
}

// ---------------------------------------------------------------------------
// WMMA GEMM: C[M,N] = A[M,Kp](bf16) @ Bt[N,Kp](bf16, transposed) (+bias)(+act)
// block = 256 threads = 8 waves; each block does a 128-row strip.
// A tile staged in LDS (<=32KB). B fragments read from global (L2-resident).
// Fragment layouts used here (A/B in contiguous k order, same for both; not the ISA table order):
//   A 16x32: lane(0..15)=row M, laneHi selects K {0-15} vs {16-31}
//   B 32x16: n = lane&15, laneHi selects K {0-15} vs {16-31} (contiguous in Bt)
//   C 16x16: m = r + 8*laneHi, n = lane&15
// ---------------------------------------------------------------------------
__global__ void __launch_bounds__(256)
gemm_bf16_wmma(const unsigned short* __restrict__ A,
               const unsigned short* __restrict__ Bt,
               const float* __restrict__ bias,
               float* __restrict__ C,
               int M, int N, int Kp, int act) {
  extern __shared__ unsigned short At[];   // [128][Kp]
  const int tid = threadIdx.x;
  const int row0 = blockIdx.x * 128;
  const int tileElems = 128 * Kp;
  for (int i = tid; i < tileElems; i += 256) {
    int r = i / Kp, k = i - r * Kp;
    int gr = row0 + r;
    At[i] = (gr < M) ? A[(size_t)gr * Kp + k] : (unsigned short)0;
  }
  __syncthreads();

  const int wave = tid >> 5, lane = tid & 31;
  const int laneHi = lane >> 4, mrow = lane & 15;
  const unsigned short* arow = At + (size_t)(wave * 16 + mrow) * Kp;
  const int kbA = laneHi * 16, kbB = laneHi * 16;

  for (int ct = 0; ct < N; ct += 16) {
    const int col = ct + mrow;
    v8f acc;
    const float b0 = bias ? bias[col] : 0.f;
#pragma unroll
    for (int r = 0; r < 8; ++r) acc[r] = b0;
    const unsigned short* brow = Bt + (size_t)col * Kp;
    for (int k0 = 0; k0 < Kp; k0 += 32) {
      v8bf alo = *(const v8bf*)(arow + k0 + kbA);
      v8bf ahi = *(const v8bf*)(arow + k0 + kbA + 8);
      v16bf av = __builtin_shufflevector(alo, ahi, 0, 1, 2, 3, 4, 5, 6, 7,
                                         8, 9, 10, 11, 12, 13, 14, 15);
      v16bf bv = *(const v16bf*)(brow + k0 + kbB);
      acc = __builtin_amdgcn_wmma_f32_16x16x32_bf16(false, av, false, bv,
                                                    (short)0, acc, false, false);
    }
    const int gr0 = row0 + wave * 16 + laneHi * 8;
#pragma unroll
    for (int r = 0; r < 8; ++r) {
      int gr = gr0 + r;
      if (gr < M) {
        float v = acc[r];
        if (act == 1) v = fmaxf(v, 0.f);
        else if (act == 2) v = (v > 0.f) ? v : (__expf(v) - 1.f);
        C[(size_t)gr * N + col] = v;
      }
    }
  }
}

// ---------------------------------------------------------------------------
// GATv2 edge-softmax passes (self loops are edges j >= E with src=dst=j-E)
// ---------------------------------------------------------------------------
__global__ void gat_init(float* __restrict__ out, const float* __restrict__ b,
                         unsigned* __restrict__ menc, float* __restrict__ s,
                         int n, int C) {
  int i = blockIdx.x * 256 + threadIdx.x;
  if (i < n) { menc[i] = 0u; s[i] = 0.f; }
  if (i < n * C) out[i] = b[i % C];
}

// wave per edge: e = leaky_relu(xl[src]+xr[dst], 0.2) . att ; segment max via atomicMax
__global__ void gat_edge_score(const float* __restrict__ xlxr, const int* __restrict__ ei,
                               const float* __restrict__ att, float* __restrict__ eb,
                               unsigned* __restrict__ menc, int E, int n, int C) {
  int j = blockIdx.x * 8 + (threadIdx.x >> 5);
  int lane = threadIdx.x & 31;
  if (j >= E + n) return;
  int src = (j < E) ? ei[j] : (j - E);
  int dst = (j < E) ? ei[E + j] : (j - E);
  const float* xl = xlxr + (size_t)src * (2 * C);
  const float* xr = xlxr + (size_t)dst * (2 * C) + C;
  float acc = 0.f;
  for (int c = lane; c < C; c += 32) {
    float v = xl[c] + xr[c];
    v = (v > 0.f) ? v : 0.2f * v;
    acc += v * att[c];
  }
#pragma unroll
  for (int o = 16; o > 0; o >>= 1) acc += __shfl_xor(acc, o, 32);
  if (lane == 0) {
    eb[j] = acc;
    atomicMax(&menc[dst], fenc(acc));
  }
}

__global__ void gat_edge_exp(const float* __restrict__ eb, const unsigned* __restrict__ menc,
                             const int* __restrict__ ei, float* __restrict__ ab,
                             float* __restrict__ s, int E, int n) {
  int j = blockIdx.x * 256 + threadIdx.x;
  if (j >= E + n) return;
  int dst = (j < E) ? ei[E + j] : (j - E);
  float a = __expf(eb[j] - fdec(menc[dst]));
  ab[j] = a;
  atomicAdd(&s[dst], a);
}

// wave per edge: out[dst] += (a/s[dst]) * xl[src]
__global__ void gat_edge_scatter(const float* __restrict__ ab, const float* __restrict__ s,
                                 const float* __restrict__ xlxr, const int* __restrict__ ei,
                                 float* __restrict__ out, int E, int n, int C) {
  int j = blockIdx.x * 8 + (threadIdx.x >> 5);
  int lane = threadIdx.x & 31;
  if (j >= E + n) return;
  int src = (j < E) ? ei[j] : (j - E);
  int dst = (j < E) ? ei[E + j] : (j - E);
  float alpha = ab[j] / s[dst];
  const float* xl = xlxr + (size_t)src * (2 * C);
  for (int c = lane; c < C; c += 32)
    atomicAdd(&out[(size_t)dst * C + c], alpha * xl[c]);
}

// ---------------------------------------------------------------------------
// Misc elementwise / pooling / small linear kernels
// ---------------------------------------------------------------------------
__global__ void elemwise_act(float* __restrict__ x, int count, int mode) {
  int i = blockIdx.x * 256 + threadIdx.x;
  if (i >= count) return;
  float v = x[i];
  x[i] = (mode == 1) ? fmaxf(v, 0.f) : ((v > 0.f) ? v : (__expf(v) - 1.f));
}

__global__ void psp_kernel(const float* __restrict__ x, const int* __restrict__ idx,
                           const float* __restrict__ w, float* __restrict__ out,
                           int n, int K, int C) {
  int i = blockIdx.x * 256 + threadIdx.x;
  if (i >= n * C) return;
  int r = i / C, c = i - r * C;
  float acc = 0.f;
  for (int k = 0; k < K; ++k)
    acc += w[r * K + k] * x[(size_t)idx[r * K + k] * C + c];
  out[i] = acc;
}

__global__ void gate_kernel(const float* __restrict__ x, const float* __restrict__ Wg,
                            const float* __restrict__ bg, float* __restrict__ gate,
                            int n, int C) {
  int i = blockIdx.x * 256 + threadIdx.x;
  if (i >= n) return;
  float acc = bg[0];
  for (int k = 0; k < C; ++k) acc += x[(size_t)i * C + k] * Wg[k];
  gate[i] = acc;
}

// block per group; 256 nodes per group; C=32
__global__ void att_agg_kernel(const float* __restrict__ x, const float* __restrict__ gate,
                               float* __restrict__ gout) {
  __shared__ float red[256];
  __shared__ float accum[32];
  int g = blockIdx.x, t = threadIdx.x;
  int node = g * 256 + t;
  float gv = gate[node];
  red[t] = gv; __syncthreads();
  for (int o = 128; o > 0; o >>= 1) { if (t < o) red[t] = fmaxf(red[t], red[t + o]); __syncthreads(); }
  float m = red[0]; __syncthreads();
  float e = __expf(gv - m);
  red[t] = e; __syncthreads();
  for (int o = 128; o > 0; o >>= 1) { if (t < o) red[t] += red[t + o]; __syncthreads(); }
  float ssum = red[0];
  if (t < 32) accum[t] = 0.f;
  __syncthreads();
  float a = e / ssum;
  for (int c = 0; c < 32; ++c) atomicAdd(&accum[c], a * x[(size_t)node * 32 + c]);
  __syncthreads();
  if (t < 32) gout[g * 32 + t] = accum[t];
}

// d = concat([h@W_pne+b_pne, gagg[pooled_batch]]) @ W_up + b_up   (per out col)
__global__ void dec_up_kernel(const float* __restrict__ h, const float* __restrict__ Wpne,
                              const float* __restrict__ bpne, const float* __restrict__ gagg,
                              const int* __restrict__ pbatch, const float* __restrict__ Wup,
                              const float* __restrict__ bup, float* __restrict__ out, int n) {
  int i = blockIdx.x * 256 + threadIdx.x;
  if (i >= n * 32) return;
  int r = i / 32, c = i - r * 32;
  float d0 = bpne[0], d1 = bpne[1];
  for (int k = 0; k < 32; ++k) {
    float hv = h[(size_t)r * 32 + k];
    d0 += hv * Wpne[k * 2 + 0];
    d1 += hv * Wpne[k * 2 + 1];
  }
  const float* gg = gagg + (size_t)pbatch[r] * 32;
  float acc = bup[c] + d0 * Wup[0 * 32 + c] + d1 * Wup[1 * 32 + c];
  for (int k = 0; k < 32; ++k) acc += gg[k] * Wup[(2 + k) * 32 + c];
  out[i] = acc;
}

__global__ void zero_f(float* __restrict__ p, int n) {
  int i = blockIdx.x * 256 + threadIdx.x;
  if (i < n) p[i] = 0.f;
}

__global__ void bn_stats(const float* __restrict__ h, float* __restrict__ sum,
                         float* __restrict__ sq, int n, int C) {
  int i = blockIdx.x * 256 + threadIdx.x;
  if (i >= n * C) return;
  float v = h[i];
  int c = i % C;
  atomicAdd(&sum[c], v);
  atomicAdd(&sq[c], v * v);
}

__global__ void bn_relu(float* __restrict__ h, const float* __restrict__ sum,
                        const float* __restrict__ sq, const float* __restrict__ g,
                        const float* __restrict__ b, int n, int C) {
  int i = blockIdx.x * 256 + threadIdx.x;
  if (i >= n * C) return;
  int c = i % C;
  float inv_n = 1.f / (float)n;
  float mu = sum[c] * inv_n;
  float var = sq[c] * inv_n - mu * mu;
  float v = (h[i] - mu) * rsqrtf(var + 1e-5f) * g[c] + b[c];
  h[i] = fmaxf(v, 0.f);
}

__global__ void lin_out(const float* __restrict__ h, const float* __restrict__ W,
                        const float* __restrict__ b, float* __restrict__ out,
                        int n, int K, int C) {
  int i = blockIdx.x * 256 + threadIdx.x;
  if (i >= n * C) return;
  int r = i / C, c = i - r * C;
  float acc = b[c];
  for (int k = 0; k < K; ++k) acc += h[(size_t)r * K + k] * W[k * C + c];
  out[i] = acc;
}

// ---------------------------------------------------------------------------
// Host orchestration
// ---------------------------------------------------------------------------
struct Scratch {
  float *h0, *h1, *xlxr, *eb, *ab, *sb, *gate, *gagg, *stats;
  unsigned* menc;
  unsigned short *bfa, *bfw;
};

static void run_gatv2(const float* hin, int n, int K,
                      const float* Wl, const float* Wr, const float* att, const float* b,
                      const int* ei, int E, int C, float* hout,
                      const Scratch& w, hipStream_t s) {
  int Kp = (K + 31) & ~31;
  int N2C = 2 * C;
  cvt_pad_bf16<<<cdiv(n * Kp, 256), 256, 0, s>>>(hin, w.bfa, n, K, Kp);
  cvt_wcat_t<<<cdiv(N2C * Kp, 256), 256, 0, s>>>(Wl, Wr, w.bfw, K, C, Kp);
  gemm_bf16_wmma<<<cdiv(n, 128), 256, (size_t)128 * Kp * 2, s>>>(
      w.bfa, w.bfw, nullptr, w.xlxr, n, N2C, Kp, 0);
  gat_init<<<cdiv(n * C, 256), 256, 0, s>>>(hout, b, w.menc, w.sb, n, C);
  int Etot = E + n;
  gat_edge_score<<<cdiv(Etot, 8), 256, 0, s>>>(w.xlxr, ei, att, w.eb, w.menc, E, n, C);
  gat_edge_exp<<<cdiv(Etot, 256), 256, 0, s>>>(w.eb, w.menc, ei, w.ab, w.sb, E, n);
  gat_edge_scatter<<<cdiv(Etot, 8), 256, 0, s>>>(w.ab, w.sb, w.xlxr, ei, hout, E, n, C);
}

extern "C" void kernel_launch(void* const* d_in, const int* in_sizes, int n_in,
                              void* d_out, int out_size, void* d_ws, size_t ws_size,
                              hipStream_t stream) {
  (void)in_sizes; (void)n_in; (void)out_size; (void)ws_size;
  // ---- inputs (insertion order, then params as sorted pytree) ----
  const float* x            = (const float*)d_in[0];
  const int*   edge_index   = (const int*)d_in[1];
  const int*   idx_pool0    = (const int*)d_in[3];
  const int*   idx_pool1    = (const int*)d_in[4];
  const int*   idx_unpool0  = (const int*)d_in[5];
  const int*   idx_unpool1  = (const int*)d_in[6];
  const float* w_pool0      = (const float*)d_in[7];
  const float* w_pool1      = (const float*)d_in[8];
  const float* w_unpool0    = (const float*)d_in[9];
  const float* w_unpool1    = (const float*)d_in[10];
  const int*   pe0          = (const int*)d_in[11];
  const int*   pe1          = (const int*)d_in[12];
  const int*   pooled_batch = (const int*)d_in[13];
  const float* W_gate = (const float*)d_in[14];
  const float* W_pne  = (const float*)d_in[15];
  const float* W_up   = (const float*)d_in[16];
  const float* b_gate = (const float*)d_in[17];
  const float* b_pne  = (const float*)d_in[18];
  const float* b_up   = (const float*)d_in[19];
  const float* bn_b   = (const float*)d_in[20];
  const float* bn_g   = (const float*)d_in[21];
  // gat param dicts flatten as Wl, Wr, att, b
  const float *dec0_Wl = (const float*)d_in[22], *dec0_Wr = (const float*)d_in[23],
              *dec0_at = (const float*)d_in[24], *dec0_b  = (const float*)d_in[25];
  const float *dec1_Wl = (const float*)d_in[26], *dec1_Wr = (const float*)d_in[27],
              *dec1_at = (const float*)d_in[28], *dec1_b  = (const float*)d_in[29];
  const float *enc0_Wl = (const float*)d_in[30], *enc0_Wr = (const float*)d_in[31],
              *enc0_at = (const float*)d_in[32], *enc0_b  = (const float*)d_in[33];
  const float *enc1_Wl = (const float*)d_in[34], *enc1_Wr = (const float*)d_in[35],
              *enc1_at = (const float*)d_in[36], *enc1_b  = (const float*)d_in[37];
  const float* gat_Wl[4]; const float* gat_Wr[4]; const float* gat_at[4]; const float* gat_b[4];
  for (int l = 0; l < 4; ++l) {
    gat_Wl[l] = (const float*)d_in[38 + 4 * l + 0];
    gat_Wr[l] = (const float*)d_in[38 + 4 * l + 1];
    gat_at[l] = (const float*)d_in[38 + 4 * l + 2];
    gat_b[l]  = (const float*)d_in[38 + 4 * l + 3];
  }
  const float* mlp_W1 = (const float*)d_in[54];
  const float* mlp_W2 = (const float*)d_in[55];
  const float* mlp_b1 = (const float*)d_in[56];
  const float* mlp_b2 = (const float*)d_in[57];
  const float *outp_Wl = (const float*)d_in[58], *outp_Wr = (const float*)d_in[59],
              *outp_at = (const float*)d_in[60], *outp_b  = (const float*)d_in[61];

  // ---- workspace carve-up ----
  char* base = (char*)d_ws;
  size_t off = 0;
  auto take = [&](size_t bytes) -> char* {
    char* p = base + off;
    off += (bytes + 255) & ~(size_t)255;
    return p;
  };
  Scratch w;
  w.h0    = (float*)take((size_t)cN0 * 128 * 4);
  w.h1    = (float*)take((size_t)cN0 * 128 * 4);
  w.xlxr  = (float*)take((size_t)cN0 * 256 * 4);
  w.bfa   = (unsigned short*)take((size_t)cN0 * 128 * 2);
  w.bfw   = (unsigned short*)take((size_t)256 * 128 * 2);
  w.eb    = (float*)take((size_t)(cE0 + cN0) * 4);
  w.ab    = (float*)take((size_t)(cE0 + cN0) * 4);
  w.menc  = (unsigned*)take((size_t)cN0 * 4);
  w.sb    = (float*)take((size_t)cN0 * 4);
  w.gate  = (float*)take((size_t)cN2 * 4);
  w.gagg  = (float*)take((size_t)cG * 32 * 4);
  w.stats = (float*)take(128 * 4);
  float* h1mlp = w.xlxr;  // reuse: xlxr is dead after out_pre layer
  hipStream_t s = stream;

  // ---- encoder: 4x GATv2 on fine graph ----
  run_gatv2(x, cN0, 3, gat_Wl[0], gat_Wr[0], gat_at[0], gat_b[0], edge_index, cE0, 128, w.h1, w, s);
  elemwise_act<<<cdiv(cN0 * 128, 256), 256, 0, s>>>(w.h1, cN0 * 128, 1);
  run_gatv2(w.h1, cN0, 128, gat_Wl[1], gat_Wr[1], gat_at[1], gat_b[1], edge_index, cE0, 128, w.h0, w, s);
  elemwise_act<<<cdiv(cN0 * 128, 256), 256, 0, s>>>(w.h0, cN0 * 128, 1);
  run_gatv2(w.h0, cN0, 128, gat_Wl[2], gat_Wr[2], gat_at[2], gat_b[2], edge_index, cE0, 128, w.h1, w, s);
  elemwise_act<<<cdiv(cN0 * 128, 256), 256, 0, s>>>(w.h1, cN0 * 128, 1);
  run_gatv2(w.h1, cN0, 128, gat_Wl[3], gat_Wr[3], gat_at[3], gat_b[3], edge_index, cE0, 128, w.h0, w, s);
  elemwise_act<<<cdiv(cN0 * 128, 256), 256, 0, s>>>(w.h0, cN0 * 128, 2);  // elu

  psp_kernel<<<cdiv(cN1 * 128, 256), 256, 0, s>>>(w.h0, idx_pool0, w_pool0, w.h1, cN1, 8, 128);
  run_gatv2(w.h1, cN1, 128, enc0_Wl, enc0_Wr, enc0_at, enc0_b, pe0, cE1, 64, w.h0, w, s);
  elemwise_act<<<cdiv(cN1 * 64, 256), 256, 0, s>>>(w.h0, cN1 * 64, 2);

  psp_kernel<<<cdiv(cN2 * 64, 256), 256, 0, s>>>(w.h0, idx_pool1, w_pool1, w.h1, cN2, 8, 64);
  run_gatv2(w.h1, cN2, 64, enc1_Wl, enc1_Wr, enc1_at, enc1_b, pe1, cE2, 32, w.h0, w, s);
  elemwise_act<<<cdiv(cN2 * 32, 256), 256, 0, s>>>(w.h0, cN2 * 32, 2);

  // ---- bottleneck: gate + attention aggregation + concat/up projection ----
  gate_kernel<<<cdiv(cN2, 256), 256, 0, s>>>(w.h0, W_gate, b_gate, w.gate, cN2, 32);
  att_agg_kernel<<<cG, 256, 0, s>>>(w.h0, w.gate, w.gagg);
  dec_up_kernel<<<cdiv(cN2 * 32, 256), 256, 0, s>>>(w.h0, W_pne, b_pne, w.gagg,
                                                    pooled_batch, W_up, b_up, w.h1, cN2);

  // ---- decoder ----
  run_gatv2(w.h1, cN2, 32, dec0_Wl, dec0_Wr, dec0_at, dec0_b, pe1, cE2, 64, w.h0, w, s);
  elemwise_act<<<cdiv(cN2 * 64, 256), 256, 0, s>>>(w.h0, cN2 * 64, 2);
  psp_kernel<<<cdiv(cN1 * 64, 256), 256, 0, s>>>(w.h0, idx_unpool0, w_unpool0, w.h1, cN1, 4, 64);
  // faithful to reference: dec_gcn1 still uses pe1 with n = N1
  run_gatv2(w.h1, cN1, 64, dec1_Wl, dec1_Wr, dec1_at, dec1_b, pe1, cE2, 128, w.h0, w, s);
  elemwise_act<<<cdiv(cN1 * 128, 256), 256, 0, s>>>(w.h0, cN1 * 128, 2);
  psp_kernel<<<cdiv(cN0 * 128, 256), 256, 0, s>>>(w.h0, idx_unpool1, w_unpool1, w.h1, cN0, 4, 128);
  run_gatv2(w.h1, cN0, 128, outp_Wl, outp_Wr, outp_at, outp_b, edge_index, cE0, 128, w.h0, w, s);

  // ---- MLP head: Linear(128->64) -> BN(train) -> ReLU -> Linear(64->3) ----
  cvt_pad_bf16<<<cdiv(cN0 * 128, 256), 256, 0, s>>>(w.h0, w.bfa, cN0, 128, 128);
  cvt_w_t<<<cdiv(64 * 128, 256), 256, 0, s>>>(mlp_W1, w.bfw, 128, 64, 128);
  gemm_bf16_wmma<<<cdiv(cN0, 128), 256, (size_t)128 * 128 * 2, s>>>(
      w.bfa, w.bfw, mlp_b1, h1mlp, cN0, 64, 128, 0);
  zero_f<<<1, 128, 0, s>>>(w.stats, 128);
  bn_stats<<<cdiv(cN0 * 64, 256), 256, 0, s>>>(h1mlp, w.stats, w.stats + 64, cN0, 64);
  bn_relu<<<cdiv(cN0 * 64, 256), 256, 0, s>>>(h1mlp, w.stats, w.stats + 64, bn_g, bn_b, cN0, 64);

  float* outf = (float*)d_out;
  lin_out<<<cdiv(cN0 * 3, 256), 256, 0, s>>>(h1mlp, mlp_W2, mlp_b2, outf, cN0, 64, 3);

  // second tuple element: edge_index passthrough (raw int32 bits)
  hipMemcpyAsync(outf + (size_t)cN0 * 3, d_in[1], (size_t)2 * cE0 * 4,
                 hipMemcpyDeviceToDevice, s);
}
